// DirectedMessagePassing_67680094650560
// MI455X (gfx1250) — compile-verified, matrix operand order checked by reading, not run on hardware
//
#include <hip/hip_runtime.h>

#define N_HIDDEN 128
#define NODE_FDIM 133
#define EDGE_FDIM 14
#define MP_DEPTH 3
#define LDSLD 136  // padded LDS row stride in halves (multiple of 8)

typedef _Float16 v16h __attribute__((ext_vector_type(16)));
typedef _Float16 v8h  __attribute__((ext_vector_type(8)));
typedef _Float16 v4h  __attribute__((ext_vector_type(4)));
typedef float    v8f  __attribute__((ext_vector_type(8)));

// ---- WMMA fragment helpers (wave32, 16x16x32 f16 -> f32) -------------------

// K-index inside a 32-wide K block for (lane, element) of a 16-bit A/B fragment
// per CDNA5 ISA 7.12.2 (16-bit A-matrix 16x32 layout).
__device__ __forceinline__ int frag_k(int lane, int e) {
  int v = e >> 1, p = e & 1;
  int g = (lane >> 4) & 1;   // half-wave selects K sub-group
  int b = v >> 2;            // VGPR block selects K 0..15 vs 16..31
  return b * 16 + g * 8 + ((v & 3) << 1) + p;
}

// Load an A fragment from a row-major f16 matrix tile (16 rows, ldh halves).
__device__ __forceinline__ v16h load_a_rm(const _Float16* base, int ldh, int kb, int lane) {
  int m = lane & 15;
  int g = (lane >> 4) & 1;
  const _Float16* r = base + (size_t)m * ldh + kb * 32 + g * 8;
  v8h lo = *(const v8h*)(r);        // K = kb*32 + g*8 .. +7
  v8h hi = *(const v8h*)(r + 16);   // K = kb*32 + 16 + g*8 .. +7
  return __builtin_shufflevector(lo, hi, 0, 1, 2, 3, 4, 5, 6, 7,
                                 8, 9, 10, 11, 12, 13, 14, 15);
}

// Load a pre-packed B fragment (32B contiguous per lane).
__device__ __forceinline__ v16h load_b(const _Float16* Bp, int kblks, int nt, int kb, int lane) {
  return ((const v16h*)Bp)[((size_t)nt * kblks + kb) * 32 + lane];
}

__device__ __forceinline__ v8f wmma16(v16h a, v16h b, v8f c) {
  return __builtin_amdgcn_wmma_f32_16x16x32_f16(false, a, false, b, (short)0, c,
                                                false, false);
}

__device__ __forceinline__ float sigm(float x) { return 1.0f / (1.0f + __expf(-x)); }

// ---- Weight packing: f32 source -> f16 WMMA-B fragments --------------------
// transposed==0: B[k][n] = src[(rowOff+k)*128 + n]   (src is [K,128] row-major)
// transposed==1: B[k][n] = src[n*128 + k]            (src is [N,128], use src^T)
__global__ void pack_b_kernel(const float* __restrict__ src, _Float16* __restrict__ dst,
                              int Ksrc, int kblks, int ntiles, int rowOff, int transposed) {
  int t = blockIdx.x * blockDim.x + threadIdx.x;
  int total = ntiles * kblks * 32;
  if (t >= total) return;
  int lane = t & 31;
  int kb = (t >> 5) % kblks;
  int nt = (t >> 5) / kblks;
  int n = nt * 16 + (lane & 15);
  v16h frag;
#pragma unroll
  for (int e = 0; e < 16; ++e) {
    int k = kb * 32 + frag_k(lane, e);
    float v = 0.0f;
    if (k < Ksrc) v = transposed ? src[(size_t)n * 128 + k]
                                 : src[(size_t)(rowOff + k) * 128 + n];
    frag[e] = (_Float16)v;
  }
  ((v16h*)dst)[((size_t)nt * kblks + kb) * 32 + lane] = frag;
}

// ---- Stage [edge_attr | x[src] | 0pad] as f16 [E,160] ----------------------
__global__ void build_xsrc_kernel(const float* __restrict__ x,
                                  const float* __restrict__ edge_attr,
                                  const int* __restrict__ eidx,
                                  _Float16* __restrict__ out, int E) {
  int t = blockIdx.x * blockDim.x + threadIdx.x;
  if (t >= E * 40) return;
  int e = t / 40;
  int c0 = (t % 40) * 4;
  int s = eidx[e];
  v4h v;
#pragma unroll
  for (int i = 0; i < 4; ++i) {
    int c = c0 + i;
    float f = 0.0f;
    if (c < EDGE_FDIM) f = edge_attr[(size_t)e * EDGE_FDIM + c];
    else if (c < EDGE_FDIM + NODE_FDIM) f = x[(size_t)s * NODE_FDIM + (c - EDGE_FDIM)];
    v[i] = (_Float16)f;
  }
  *(v4h*)(out + (size_t)e * 160 + c0) = v;
}

// ---- Generic WMMA GEMM: out[M,128] = act(A[M,K] @ Bp + bias) ---------------
__global__ void __launch_bounds__(128) gemm128_kernel(
    const _Float16* __restrict__ A, int lda, int kblks,
    const _Float16* __restrict__ Bp, const float* __restrict__ bias,
    float* __restrict__ outF, _Float16* __restrict__ outH,
    int relu, int mtiles) {
  int wl = threadIdx.x >> 5;
  int lane = threadIdx.x & 31;
  int tile = blockIdx.x * 4 + wl;
  if (tile >= mtiles) return;
  const _Float16* Arow = A + (size_t)tile * 16 * lda;
  int m0 = (lane >> 4) * 8;
  int nn = lane & 15;
  for (int nt = 0; nt < 8; ++nt) {
    v8f acc = {};
    for (int kb = 0; kb < kblks; ++kb)
      acc = wmma16(load_a_rm(Arow, lda, kb, lane), load_b(Bp, kblks, nt, kb, lane), acc);
    float bv = bias[nt * 16 + nn];
#pragma unroll
    for (int r = 0; r < 8; ++r) {
      float v = acc[r] + bv;
      if (relu) v = v > 0.0f ? v : 0.0f;
      size_t off = (size_t)(tile * 16 + m0 + r) * N_HIDDEN + nt * 16 + nn;
      if (outF) outF[off] = v;
      if (outH) outH[off] = (_Float16)v;
    }
  }
}

// ---- Fused per-depth edge update: Linear+ReLU+Linear+GRUCell ---------------
// A fragments are hoisted into registers: aM (messages), aI (mi), aN (nm) are
// invariant across the N-tile / gate loops, so the GRU inner loop issues only
// packed-B fragment loads (L2-resident weights) between back-to-back WMMAs.
__global__ void __launch_bounds__(128) edge_step_kernel(
    const float* __restrict__ pre1,
    float* __restrict__ msgF, _Float16* __restrict__ msgH,
    const _Float16* __restrict__ Bm1m, const _Float16* __restrict__ Bm2,
    const _Float16* __restrict__ Bih, const _Float16* __restrict__ Bhh,
    const float* __restrict__ bm2, const float* __restrict__ b_ih,
    const float* __restrict__ b_hh, int mtiles) {
  __shared__ _Float16 smi[4][16 * LDSLD];
  __shared__ _Float16 snm[4][16 * LDSLD];
  const int wl = threadIdx.x >> 5;
  const int lane = threadIdx.x & 31;
  const int tile = blockIdx.x * 4 + wl;
  if (tile >= mtiles) return;
  const size_t base = (size_t)tile * 16;
  const int m0 = (lane >> 4) * 8;
  const int nn = lane & 15;
  _Float16* SI = smi[wl];
  _Float16* SN = snm[wl];

  // Hoist A fragments of the current messages straight from global row-major
  // (all reads complete before the in-place writes in the GRU stage).
  const _Float16* Mrow = msgH + base * N_HIDDEN;
  v16h aM[4];
#pragma unroll
  for (int kb = 0; kb < 4; ++kb) aM[kb] = load_a_rm(Mrow, N_HIDDEN, kb, lane);

  // mi = relu(pre1 + msg @ Wm1[147:275])  -> SI (LDS, for cross-lane retile)
  for (int nt = 0; nt < 8; ++nt) {
    v8f acc = {};
#pragma unroll
    for (int kb = 0; kb < 4; ++kb)
      acc = wmma16(aM[kb], load_b(Bm1m, 4, nt, kb, lane), acc);
#pragma unroll
    for (int r = 0; r < 8; ++r) {
      float v = acc[r] + pre1[(base + m0 + r) * N_HIDDEN + nt * 16 + nn];
      v = v > 0.0f ? v : 0.0f;
      SI[(m0 + r) * LDSLD + nt * 16 + nn] = (_Float16)v;
    }
  }
  v16h aI[4];
#pragma unroll
  for (int kb = 0; kb < 4; ++kb) aI[kb] = load_a_rm(SI, LDSLD, kb, lane);

  // nm = mi @ Wm2 + bm2  -> SN (LDS retile)
  for (int nt = 0; nt < 8; ++nt) {
    v8f acc = {};
#pragma unroll
    for (int kb = 0; kb < 4; ++kb)
      acc = wmma16(aI[kb], load_b(Bm2, 4, nt, kb, lane), acc);
    float bv = bm2[nt * 16 + nn];
#pragma unroll
    for (int r = 0; r < 8; ++r)
      SN[(m0 + r) * LDSLD + nt * 16 + nn] = (_Float16)(acc[r] + bv);
  }
  v16h aN[4];
#pragma unroll
  for (int kb = 0; kb < 4; ++kb) aN[kb] = load_a_rm(SN, LDSLD, kb, lane);

  // GRUCell: r/z gates accumulate nm@Wih^T and msg@Whh^T into one accumulator.
  for (int t = 0; t < 8; ++t) {
    v8f aR = {}, aZ = {}, aGi = {}, aGh = {};
#pragma unroll
    for (int kb = 0; kb < 4; ++kb) {
      aR = wmma16(aN[kb], load_b(Bih, 4, t, kb, lane), aR);
      aR = wmma16(aM[kb], load_b(Bhh, 4, t, kb, lane), aR);
      aZ = wmma16(aN[kb], load_b(Bih, 4, 8 + t, kb, lane), aZ);
      aZ = wmma16(aM[kb], load_b(Bhh, 4, 8 + t, kb, lane), aZ);
      aGi = wmma16(aN[kb], load_b(Bih, 4, 16 + t, kb, lane), aGi);
      aGh = wmma16(aM[kb], load_b(Bhh, 4, 16 + t, kb, lane), aGh);
    }
    int col = t * 16 + nn;
    float bR = b_ih[col] + b_hh[col];
    float bZ = b_ih[128 + col] + b_hh[128 + col];
    float bGi = b_ih[256 + col];
    float bGh = b_hh[256 + col];
#pragma unroll
    for (int r = 0; r < 8; ++r) {
      float rr = sigm(aR[r] + bR);
      float zz = sigm(aZ[r] + bZ);
      float nv = tanhf((aGi[r] + bGi) + rr * (aGh[r] + bGh));
      size_t off = (base + m0 + r) * N_HIDDEN + col;
      float h = msgF[off];
      float hn = (1.0f - zz) * nv + zz * h;
      msgF[off] = hn;
      msgH[off] = (_Float16)hn;
    }
  }
}

// ---- segment_sum via float atomics -----------------------------------------
__global__ void scatter_add_kernel(const float* __restrict__ msgF,
                                   const int* __restrict__ eidx,
                                   float* __restrict__ nodeM, int E) {
  int t = blockIdx.x * blockDim.x + threadIdx.x;
  if (t >= E * 32) return;
  int e = t >> 5;
  int c0 = (t & 31) << 2;
  int tg = eidx[E + e];
#pragma unroll
  for (int i = 0; i < 4; ++i)
    atomicAdd(&nodeM[(size_t)tg * N_HIDDEN + c0 + i],
              msgF[(size_t)e * N_HIDDEN + c0 + i]);
}

// ---- Stage [x | node_messages | 0pad] as f16 [N,288] -----------------------
__global__ void build_nodeA_kernel(const float* __restrict__ x,
                                   const float* __restrict__ nodeM,
                                   _Float16* __restrict__ out, int Nn) {
  int t = blockIdx.x * blockDim.x + threadIdx.x;
  if (t >= Nn * 72) return;
  int n = t / 72;
  int c0 = (t % 72) * 4;
  v4h v;
#pragma unroll
  for (int i = 0; i < 4; ++i) {
    int c = c0 + i;
    float f = 0.0f;
    if (c < NODE_FDIM) f = x[(size_t)n * NODE_FDIM + c];
    else if (c < NODE_FDIM + N_HIDDEN) f = nodeM[(size_t)n * N_HIDDEN + (c - NODE_FDIM)];
    v[i] = (_Float16)f;
  }
  *(v4h*)(out + (size_t)n * 288 + c0) = v;
}

// ---------------------------------------------------------------------------

static inline int cdiv(long long a, long long b) { return (int)((a + b - 1) / b); }

extern "C" void kernel_launch(void* const* d_in, const int* in_sizes, int n_in,
                              void* d_out, int out_size, void* d_ws, size_t ws_size,
                              hipStream_t stream) {
  (void)n_in; (void)out_size; (void)ws_size;
  const float* x         = (const float*)d_in[0];
  const float* edge_attr = (const float*)d_in[1];
  const int*   eidx      = (const int*)d_in[2];
  const float* Wm1 = (const float*)d_in[3];  const float* bm1 = (const float*)d_in[4];
  const float* Wm2 = (const float*)d_in[5];  const float* bm2 = (const float*)d_in[6];
  const float* Wih = (const float*)d_in[7];  const float* bih = (const float*)d_in[8];
  const float* Whh = (const float*)d_in[9];  const float* bhh = (const float*)d_in[10];
  const float* Wn  = (const float*)d_in[11]; const float* bn  = (const float*)d_in[12];
  const float* Wo1 = (const float*)d_in[13]; const float* bo1 = (const float*)d_in[14];
  const float* Wo2 = (const float*)d_in[15]; const float* bo2 = (const float*)d_in[16];
  float* out = (float*)d_out;

  const int E  = in_sizes[1] / EDGE_FDIM;   // 400000
  const int Nn = in_sizes[0] / NODE_FDIM;   // 100000
  const int etiles = E / 16;                // 25000
  const int ntilesN = (Nn + 15) / 16;       // 6250

  // -------- workspace carve-out --------
  char* ws = (char*)d_ws;
  auto take = [&](size_t bytes) {
    char* p = ws;
    ws += (bytes + 255) & ~(size_t)255;
    return p;
  };
  _Float16* BM1A = (_Float16*)take((size_t)8 * 5 * 32 * 16 * 2);   // Wm1[0:147] K-pad 160
  _Float16* BM1M = (_Float16*)take((size_t)8 * 4 * 32 * 16 * 2);   // Wm1[147:275]
  _Float16* BM2  = (_Float16*)take((size_t)8 * 4 * 32 * 16 * 2);
  _Float16* BIH  = (_Float16*)take((size_t)24 * 4 * 32 * 16 * 2);  // W_ih^T
  _Float16* BHH  = (_Float16*)take((size_t)24 * 4 * 32 * 16 * 2);  // W_hh^T
  _Float16* BWN  = (_Float16*)take((size_t)8 * 9 * 32 * 16 * 2);   // Wn K-pad 288
  _Float16* BO1  = (_Float16*)take((size_t)8 * 4 * 32 * 16 * 2);
  _Float16* BO2  = (_Float16*)take((size_t)8 * 4 * 32 * 16 * 2);
  _Float16* xsrcA = (_Float16*)take((size_t)E * 160 * 2);
  float*    pre1  = (float*)take((size_t)E * N_HIDDEN * 4);
  float*    msgF  = (float*)take((size_t)E * N_HIDDEN * 4);
  _Float16* msgH  = (_Float16*)take((size_t)E * N_HIDDEN * 2);
  float*    nodeM = (float*)take((size_t)Nn * N_HIDDEN * 4);
  _Float16* nodeA = (_Float16*)take((size_t)Nn * 288 * 2);
  _Float16* repH  = (_Float16*)take((size_t)Nn * N_HIDDEN * 2);
  _Float16* tH    = (_Float16*)take((size_t)Nn * N_HIDDEN * 2);

  // -------- pack weights into WMMA B fragments --------
  auto pack = [&](const float* src, _Float16* dst, int K, int kblks, int ntl,
                  int rowOff, int tr) {
    int total = ntl * kblks * 32;
    pack_b_kernel<<<cdiv(total, 256), 256, 0, stream>>>(src, dst, K, kblks, ntl, rowOff, tr);
  };
  pack(Wm1, BM1A, 147, 5, 8, 0, 0);
  pack(Wm1, BM1M, 128, 4, 8, 147, 0);
  pack(Wm2, BM2, 128, 4, 8, 0, 0);
  pack(Wih, BIH, 128, 4, 24, 0, 1);
  pack(Whh, BHH, 128, 4, 24, 0, 1);
  pack(Wn, BWN, 261, 9, 8, 0, 0);
  pack(Wo1, BO1, 128, 4, 8, 0, 0);
  pack(Wo2, BO2, 128, 4, 8, 0, 0);

  // -------- edge-side staging + depth-invariant pre1 --------
  hipMemsetAsync(msgF, 0, (size_t)E * N_HIDDEN * 4, stream);
  hipMemsetAsync(msgH, 0, (size_t)E * N_HIDDEN * 2, stream);
  build_xsrc_kernel<<<cdiv((long long)E * 40, 256), 256, 0, stream>>>(
      x, edge_attr, eidx, xsrcA, E);
  gemm128_kernel<<<cdiv(etiles, 4), 128, 0, stream>>>(
      xsrcA, 160, 5, BM1A, bm1, pre1, nullptr, 0, etiles);

  // -------- DEPTH fused message-passing steps --------
  for (int d = 0; d < MP_DEPTH; ++d) {
    edge_step_kernel<<<cdiv(etiles, 4), 128, 0, stream>>>(
        pre1, msgF, msgH, BM1M, BM2, BIH, BHH, bm2, bih, bhh, etiles);
  }

  // -------- aggregate to nodes + node MLP --------
  hipMemsetAsync(nodeM, 0, (size_t)Nn * N_HIDDEN * 4, stream);
  scatter_add_kernel<<<cdiv((long long)E * 32, 256), 256, 0, stream>>>(msgF, eidx, nodeM, E);
  build_nodeA_kernel<<<cdiv((long long)Nn * 72, 256), 256, 0, stream>>>(x, nodeM, nodeA, Nn);
  gemm128_kernel<<<cdiv(ntilesN, 4), 128, 0, stream>>>(
      nodeA, 288, 9, BWN, bn, nullptr, repH, 0, ntilesN);
  gemm128_kernel<<<cdiv(ntilesN, 4), 128, 0, stream>>>(
      repH, 128, 4, BO1, bo1, nullptr, tH, 1, ntilesN);
  gemm128_kernel<<<cdiv(ntilesN, 4), 128, 0, stream>>>(
      tH, 128, 4, BO2, bo2, out, nullptr, 0, ntilesN);
}
